// VGG3_Test_B_23424751633159
// MI455X (gfx1250) — matrix operand order repaired and checked by reading, not compiled, not run on hardware
//
#include <hip/hip_runtime.h>

#define EPS 1e-5f

typedef __attribute__((ext_vector_type(16))) _Float16 v16h;
typedef __attribute__((ext_vector_type(8)))  float    v8f;

// ---------------- gfx1250 async global->LDS path (guarded) ----------------
#if defined(__AMDGCN__) && \
    __has_builtin(__builtin_amdgcn_global_load_async_to_lds_b128) && \
    __has_builtin(__builtin_amdgcn_s_wait_asynccnt)
#define USE_ASYNC_LDS 1
typedef int v4i_b128 __attribute__((vector_size(4 * sizeof(int))));
typedef __attribute__((address_space(1))) v4i_b128* gv4p;   // global int4*
typedef __attribute__((address_space(3))) v4i_b128* lv4p;   // LDS int4*
// copy n4 16-byte packets, lane-strided, wave-wide; waits ASYNCcnt==0
__device__ __forceinline__ void async_copy_f4(const float* src, float* lds_dst,
                                              int n4, int lane) {
  for (int i4 = lane; i4 < n4; i4 += 32) {
    __builtin_amdgcn_global_load_async_to_lds_b128(
        (gv4p)(void*)(src + 4 * i4),
        (lv4p)(void*)(lds_dst + 4 * i4),
        0, 0);
  }
  __builtin_amdgcn_s_wait_asynccnt(0);
}
#else
#define USE_ASYNC_LDS 0
#endif

__device__ __forceinline__ void stage_to_lds(const float* src, float* lds_dst,
                                             int n_floats, int lane) {
#if USE_ASYNC_LDS
  async_copy_f4(src, lds_dst, n_floats >> 2, lane);   // n_floats multiple of 4
#else
  for (int i = lane; i < n_floats; i += 32) lds_dst[i] = src[i];
#endif
}

// ---------------- reduction helper (wave32) ----------------
__device__ __forceinline__ float warp_sum(float v) {
#pragma unroll
  for (int o = 16; o > 0; o >>= 1) v += __shfl_xor(v, o, 32);
  return v;
}

// ---------------- K0: partial sums of x (global mean) ----------------
__global__ __launch_bounds__(256) void k_sum_x(const float* __restrict__ x,
                                               float* __restrict__ P0, int n) {
  float s = 0.f;
  const int stride = gridDim.x * blockDim.x;
  for (int i = blockIdx.x * blockDim.x + threadIdx.x; i < n; i += stride) {
    if (i + stride < n) __builtin_prefetch(&x[i + stride], 0, 0);  // global_prefetch_b8
    s += x[i];
  }
  s = warp_sum(s);
  __shared__ float red[8];
  if ((threadIdx.x & 31) == 0) red[threadIdx.x >> 5] = s;
  __syncthreads();
  if (threadIdx.x == 0) {
    float t = 0.f;
    for (int w = 0; w < 8; ++w) t += red[w];
    P0[blockIdx.x] = t;
  }
}

// ---------------- K0b: finalize mean (scaled by 100) ----------------
__global__ __launch_bounds__(256) void k_finalize_mean(const float* __restrict__ P0,
                                                       int np, float invN,
                                                       float* __restrict__ S) {
  float s = 0.f;
  for (int i = threadIdx.x; i < np; i += blockDim.x) s += P0[i];
  s = warp_sum(s);
  __shared__ float red[8];
  if ((threadIdx.x & 31) == 0) red[threadIdx.x >> 5] = s;
  __syncthreads();
  if (threadIdx.x == 0) {
    float t = 0.f;
    for (int w = 0; w < 8; ++w) t += red[w];
    S[0] = 100.f * t * invN;   // mean of (x*100)
  }
}

// ---------------- finalize BN affine: a = g*rsqrt(var+eps), c = be - mean*a ----
__global__ __launch_bounds__(256) void k_finalize_bn(const float* __restrict__ P,
                                                     int npairs, float invCnt,
                                                     const float* __restrict__ g,
                                                     const float* __restrict__ be,
                                                     float* __restrict__ S, int slot) {
  float s = 0.f, q = 0.f;
  for (int i = threadIdx.x; i < npairs; i += blockDim.x) {
    s += P[2 * i];
    q += P[2 * i + 1];
  }
  s = warp_sum(s);
  q = warp_sum(q);
  __shared__ float rs[8], rq[8];
  if ((threadIdx.x & 31) == 0) { rs[threadIdx.x >> 5] = s; rq[threadIdx.x >> 5] = q; }
  __syncthreads();
  if (threadIdx.x == 0) {
    float ts = 0.f, tq = 0.f;
    for (int w = 0; w < 8; ++w) { ts += rs[w]; tq += rq[w]; }
    float mean = ts * invCnt;
    float var  = tq * invCnt - mean * mean;
    float a = g[0] * rsqrtf(var + EPS);
    S[slot]     = a;
    S[slot + 1] = be[0] - mean * a;
  }
}

// ---------------- K1: center, conv1(sign w1)+b1, maxpool 28->14, stats --------
// one wave per image, 8 images per block; image staged via async-to-LDS
__global__ __launch_bounds__(256) void k_conv1(const float* __restrict__ x,
                                               const float* __restrict__ w1,
                                               const float* __restrict__ b1,
                                               const float* __restrict__ S,
                                               float* __restrict__ pooled1,
                                               float* __restrict__ P1, int B) {
  __shared__ __align__(16) float raw[8][784];  // async landing zone (28x28)
  __shared__ float lds[8][900];                // 30x30 zero-padded centered image
  __shared__ float redS[8], redQ[8];
  const int warp = threadIdx.x >> 5, lane = threadIdx.x & 31;
  const int img  = blockIdx.x * 8 + warp;
  const bool act = img < B;
  const float mu = S[0];
  float sw[9];
#pragma unroll
  for (int k = 0; k < 9; ++k) sw[k] = (w1[k] >= 0.f) ? 1.f : -1.f;
  float* L = lds[warp];
  for (int i = lane; i < 900; i += 32) L[i] = 0.f;
  if (act) stage_to_lds(x + (size_t)img * 784, raw[warp], 784, lane);
  __syncthreads();
  if (act) {
    for (int i = lane; i < 784; i += 32) {
      int r = i / 28, c = i - r * 28;
      L[(r + 1) * 30 + (c + 1)] = 100.f * raw[warp][i] - mu;
    }
  }
  __syncthreads();
  float s = 0.f, q = 0.f;
  if (act) {
    const float bias = b1[0];
    float* out = pooled1 + (size_t)img * 196;
    for (int p = lane; p < 196; p += 32) {
      int py = p / 14, px = p - py * 14;
      float m = -3.402823466e+38f;
#pragma unroll
      for (int dy = 0; dy < 2; ++dy)
#pragma unroll
        for (int dx = 0; dx < 2; ++dx) {
          int oy = 2 * py + dy, ox = 2 * px + dx;
          float acc = 0.f;
#pragma unroll
          for (int ky = 0; ky < 3; ++ky)
#pragma unroll
            for (int kx = 0; kx < 3; ++kx)
              acc += sw[ky * 3 + kx] * L[(oy + ky) * 30 + (ox + kx)];
          m = fmaxf(m, acc);
        }
      float v = m + bias;
      out[p] = v;
      s += v; q += v * v;
    }
  }
  s = warp_sum(s); q = warp_sum(q);
  if (lane == 0) { redS[warp] = s; redQ[warp] = q; }
  __syncthreads();
  if (threadIdx.x == 0) {
    float ts = 0.f, tq = 0.f;
    for (int w = 0; w < 8; ++w) { ts += redS[w]; tq += redQ[w]; }
    P1[2 * blockIdx.x] = ts; P1[2 * blockIdx.x + 1] = tq;
  }
}

// ---------------- K2: bn1+relu+binarize, conv2+b2, maxpool 14->7, stats -------
__global__ __launch_bounds__(256) void k_conv2(const float* __restrict__ pooled1,
                                               const float* __restrict__ w2,
                                               const float* __restrict__ b2,
                                               const float* __restrict__ S,
                                               float* __restrict__ pooled2,
                                               float* __restrict__ P2, int B) {
  __shared__ __align__(16) float raw[8][196];  // async landing zone (14x14)
  __shared__ float lds[8][256];                // 16x16 zero-padded binarized map
  __shared__ float redS[8], redQ[8];
  const int warp = threadIdx.x >> 5, lane = threadIdx.x & 31;
  const int img  = blockIdx.x * 8 + warp;
  const bool act = img < B;
  const float a1 = S[1], c1 = S[2];
  float sw[9];
#pragma unroll
  for (int k = 0; k < 9; ++k) sw[k] = (w2[k] >= 0.f) ? 1.f : -1.f;
  float* L = lds[warp];
  for (int i = lane; i < 256; i += 32) L[i] = 0.f;
  if (act) stage_to_lds(pooled1 + (size_t)img * 196, raw[warp], 196, lane);
  __syncthreads();
  if (act) {
    for (int i = lane; i < 196; i += 32) {
      int r = i / 14, c = i - r * 14;
      float v  = a1 * raw[warp][i] + c1;
      float rl = fmaxf(v, 0.f);
      L[(r + 1) * 16 + (c + 1)] = (rl >= 0.f) ? 1.f : -1.f;
    }
  }
  __syncthreads();
  float s = 0.f, q = 0.f;
  if (act) {
    const float bias = b2[0];
    float* out = pooled2 + (size_t)img * 49;
    for (int p = lane; p < 49; p += 32) {
      int py = p / 7, px = p - py * 7;
      float m = -3.402823466e+38f;
#pragma unroll
      for (int dy = 0; dy < 2; ++dy)
#pragma unroll
        for (int dx = 0; dx < 2; ++dx) {
          int oy = 2 * py + dy, ox = 2 * px + dx;
          float acc = 0.f;
#pragma unroll
          for (int ky = 0; ky < 3; ++ky)
#pragma unroll
            for (int kx = 0; kx < 3; ++kx)
              acc += sw[ky * 3 + kx] * L[(oy + ky) * 16 + (ox + kx)];
          m = fmaxf(m, acc);
        }
      float v = m + bias;
      out[p] = v;
      s += v; q += v * v;
    }
  }
  s = warp_sum(s); q = warp_sum(q);
  if (lane == 0) { redS[warp] = s; redQ[warp] = q; }
  __syncthreads();
  if (threadIdx.x == 0) {
    float ts = 0.f, tq = 0.f;
    for (int w = 0; w < 8; ++w) { ts += redS[w]; tq += redQ[w]; }
    P2[2 * blockIdx.x] = ts; P2[2 * blockIdx.x + 1] = tq;
  }
}

// ---------------- K3: FC1 via WMMA f32_16x16x32_f16 ----------------
// One wave = 16 rows. Wave stages its 16x49-float row tile (784 contiguous,
// 16B-aligned floats) into LDS via async-to-LDS, then builds A from LDS.
// K=49 padded to 64 -> two chained K=32 WMMAs. B-matrix: sign(wf1[k])
// replicated across all 16 N columns, so any column is the GEMV result;
// extract from lanes 0/16 per the documented 16x16 f32 C/D layout.
__global__ __launch_bounds__(256) void k_fc1_wmma(const float* __restrict__ pooled2,
                                                  const float* __restrict__ wf1,
                                                  const float* __restrict__ bf1,
                                                  const float* __restrict__ S,
                                                  float* __restrict__ t,
                                                  float* __restrict__ P3) {
  __shared__ __align__(16) float raw[8][784];  // 16 rows x 49 floats per wave
  const int warp = threadIdx.x >> 5, lane = threadIdx.x & 31;
  const int base = (blockIdx.x * 8 + warp) * 16;
  const bool hi  = lane >= 16;
  const int rloc = lane & 15;
  const float a2 = S[3], c2 = S[4];

  stage_to_lds(pooled2 + (size_t)base * 49, raw[warp], 784, lane);
  __syncthreads();

  const float* R = raw[warp] + rloc * 49;
  v16h A0, A1, B0, B1;
#pragma unroll
  for (int e = 0; e < 16; ++e) {
    // A layout (16-bit 16x32): lane<16 holds K={0..7,16..23}, lane>=16 K={8..15,24..31}
    int k0 = ((e < 8) ? e : e + 8) + (hi ? 8 : 0);
    int k1 = k0 + 32;
    float u0 = 0.f, u1 = 0.f;
    if (k0 < 49) {
      float v = a2 * R[k0] + c2;
      float r = fmaxf(v, 0.f);
      u0 = (r >= 0.f) ? 1.f : -1.f;
    }
    if (k1 < 49) {
      float v = a2 * R[k1] + c2;
      float r = fmaxf(v, 0.f);
      u1 = (r >= 0.f) ? 1.f : -1.f;
    }
    A0[e] = (_Float16)u0;
    A1[e] = (_Float16)u1;
    // B layout (16-bit 32x16): lane<16 holds K={0..7,16..23}, lane>=16 K={8..15,24..31} (N = lane%16)
    int kb0 = ((e < 8) ? e : e + 8) + (hi ? 8 : 0);
    int kb1 = kb0 + 32;
    B0[e] = (_Float16)((kb0 < 49) ? ((wf1[kb0] >= 0.f) ? 1.f : -1.f) : 0.f);
    B1[e] = (_Float16)((kb1 < 49) ? ((wf1[kb1] >= 0.f) ? 1.f : -1.f) : 0.f);
  }

  v8f C;
  const float bias = bf1[0];
#pragma unroll
  for (int r = 0; r < 8; ++r) C[r] = bias;

  C = __builtin_amdgcn_wmma_f32_16x16x32_f16(false, A0, false, B0, (short)0, C, false, false);
  C = __builtin_amdgcn_wmma_f32_16x16x32_f16(false, A1, false, B1, (short)0, C, false, false);

  // C/D layout: lanes 0-15 VGPR r -> M=r ; lanes 16-31 VGPR r -> M=r+8
  float s = 0.f, q = 0.f;
  if (lane == 0 || lane == 16) {
    float* tp = t + base + (hi ? 8 : 0);
#pragma unroll
    for (int r = 0; r < 8; ++r) { float v = C[r]; tp[r] = v; s += v; q += v * v; }
  }
  s += __shfl_xor(s, 16, 32);
  q += __shfl_xor(q, 16, 32);
  __shared__ float rs[8], rq[8];
  if (lane == 0) { rs[warp] = s; rq[warp] = q; }
  __syncthreads();
  if (threadIdx.x == 0) {
    float ts = 0.f, tq = 0.f;
    for (int w = 0; w < 8; ++w) { ts += rs[w]; tq += rq[w]; }
    P3[2 * blockIdx.x] = ts; P3[2 * blockIdx.x + 1] = tq;
  }
}

// ---------------- K4: bn1d+relu+binarize, final binarized linear, scale -------
__global__ __launch_bounds__(256) void k_final(const float* __restrict__ t,
                                               const float* __restrict__ S,
                                               const float* __restrict__ wf2,
                                               const float* __restrict__ bf2,
                                               const float* __restrict__ scale,
                                               float* __restrict__ out, int n) {
  int i = blockIdx.x * blockDim.x + threadIdx.x;
  if (i >= n) return;
  float v  = S[5] * t[i] + S[6];
  float r  = fmaxf(v, 0.f);
  float u  = (r >= 0.f) ? 1.f : -1.f;
  float sw = (wf2[0] >= 0.f) ? 1.f : -1.f;
  out[i] = (u * sw + bf2[0]) * scale[0];
}

// ---------------- host launcher ----------------
extern "C" void kernel_launch(void* const* d_in, const int* in_sizes, int n_in,
                              void* d_out, int out_size, void* d_ws, size_t ws_size,
                              hipStream_t stream) {
  const float* x   = (const float*)d_in[0];
  const float* w1  = (const float*)d_in[1];
  const float* b1  = (const float*)d_in[2];
  const float* g1  = (const float*)d_in[3];
  const float* be1 = (const float*)d_in[4];
  const float* w2  = (const float*)d_in[5];
  const float* b2  = (const float*)d_in[6];
  const float* g2  = (const float*)d_in[7];
  const float* be2 = (const float*)d_in[8];
  const float* wf1 = (const float*)d_in[9];
  const float* bf1 = (const float*)d_in[10];
  const float* g3  = (const float*)d_in[11];
  const float* be3 = (const float*)d_in[12];
  const float* wf2 = (const float*)d_in[13];
  const float* bf2 = (const float*)d_in[14];
  const float* scl = (const float*)d_in[15];
  float* outp = (float*)d_out;

  const int B   = in_sizes[0] / 784;   // 65536
  const int n_x = in_sizes[0];

  const int NP0 = 4096;
  const int NB1 = (B + 7) / 8;         // conv1 blocks (8 imgs each)
  const int NB2 = (B + 7) / 8;         // conv2 blocks
  const int NBF = B / 128;             // fc1 blocks (8 waves x 16 rows)

  float* S       = (float*)d_ws;                 // 16 scalar slots
  float* P0      = S + 16;                       // NP0
  float* P1      = P0 + NP0;                     // 2*NB1
  float* P2      = P1 + 2 * NB1;                 // 2*NB2
  float* P3      = P2 + 2 * NB2;                 // 2*NBF
  float* pooled1 = P3 + 2 * NBF;                 // B*196
  float* pooled2 = pooled1 + (size_t)B * 196;    // B*49
  float* tbuf    = pooled2 + (size_t)B * 49;     // B

  k_sum_x        <<<NP0, 256, 0, stream>>>(x, P0, n_x);
  k_finalize_mean<<<1, 256, 0, stream>>>(P0, NP0, 1.f / (float)n_x, S);
  k_conv1        <<<NB1, 256, 0, stream>>>(x, w1, b1, S, pooled1, P1, B);
  k_finalize_bn  <<<1, 256, 0, stream>>>(P1, NB1, 1.f / ((float)B * 196.f), g1, be1, S, 1);
  k_conv2        <<<NB2, 256, 0, stream>>>(pooled1, w2, b2, S, pooled2, P2, B);
  k_finalize_bn  <<<1, 256, 0, stream>>>(P2, NB2, 1.f / ((float)B * 49.f), g2, be2, S, 3);
  k_fc1_wmma     <<<NBF, 256, 0, stream>>>(pooled2, wf1, bf1, S, tbuf, P3);
  k_finalize_bn  <<<1, 256, 0, stream>>>(P3, NBF, 1.f / (float)B, g3, be3, S, 5);
  k_final        <<<(B + 255) / 256, 256, 0, stream>>>(tbuf, S, wf2, bf2, scl, outp, B);

  (void)n_in; (void)out_size; (void)ws_size;
}
